// GraphAttn_34875134444207
// MI455X (gfx1250) — matrix operand order repaired and checked by reading, not compiled, not run on hardware
//
#include <hip/hip_runtime.h>

// ---------------------------------------------------------------------------
// GraphAttn for MI455X (gfx1250, wave32, WMMA)
//   h        = hidden @ W_w^T + W_b                  (B,N,D)
//   w[j]     = exp(h[j]. a2 + a_b)                   (e_i term cancels in softmax)
//   S[i,j]   = mask[i,j] * w[j]
//   h'[i,:]  = (S @ h)[i,:] / (S @ ones)[i]
// Main matmul: v_wmma_f32_16x16x32_bf16 (f32 accum), single pass over j.
// hT staged to LDS via global_load_async_to_lds_b128 (triple buffer, counted
// s_wait_asynccnt); sched_group_barrier pins fragment loads 2 tiles ahead of
// their WMMAs; mask streamed non-temporally + prefetched.
// ---------------------------------------------------------------------------

typedef __attribute__((ext_vector_type(16))) __bf16 v16bf;
typedef __attribute__((ext_vector_type(2)))  __bf16 v2bf;
typedef __attribute__((ext_vector_type(8)))  float  v8f;
typedef __attribute__((ext_vector_type(4)))  float  f4;
typedef __attribute__((ext_vector_type(4)))  unsigned int u4;
typedef __attribute__((ext_vector_type(2)))  unsigned int u2;
typedef __attribute__((ext_vector_type(4)))  int     i4;

#define B_ 2
#define N_ 4096
#define D_ 256

#define K2WAVES 4                 // waves per block in kernel 2
#define ROW_BYTES 96              // padded LDS row (48 bf16) for bank spread
#define BUFB (D_ * ROW_BYTES)     // 24576 bytes per LDS buffer
#define NBUF 3                    // triple-buffered staging

#if __has_builtin(__builtin_amdgcn_global_load_async_to_lds_b128)
#define USE_ASYNC_STAGE 1
typedef __attribute__((address_space(1))) i4 as1_i4;   // global int4
typedef __attribute__((address_space(3))) i4 as3_i4;   // LDS int4
#if __has_builtin(__builtin_amdgcn_s_wait_asynccnt)
#define WAIT_ASYNC(n) __builtin_amdgcn_s_wait_asynccnt(n)
#else
#define WAIT_ASYNC(n) asm volatile("s_wait_asynccnt %0" ::"i"(n) : "memory")
#endif
#else
#define WAIT_ASYNC(n) ((void)0)
#endif

// scheduling-group directive (immediate args required -> template)
// masks: 0x008 = MFMA/WMMA, 0x020 = VMEM read, 0x100 = DS read
template <int M, int S>
__device__ __forceinline__ void sgb() {
#if __has_builtin(__builtin_amdgcn_sched_group_barrier)
  __builtin_amdgcn_sched_group_barrier(M, S, 0);
#endif
}

// request: 4 fragment loads up front, then (1 WMMA : 2 loads) so loads run
// two tiles ahead of the consuming WMMA; TAIL extra WMMAs at the end.
template <int LDMASK, int TAIL>
__device__ __forceinline__ void pipeline_hint() {
#if __has_builtin(__builtin_amdgcn_sched_group_barrier)
  sgb<LDMASK, 4>();
#pragma unroll
  for (int t = 0; t < 14; ++t) {
    sgb<0x008, 1>();
    sgb<LDMASK, 2>();
  }
  sgb<0x008, TAIL>();
#endif
}

// packed f32 -> bf16 convert (one VALU op per 2 elements)
#if __has_builtin(__builtin_amdgcn_cvt_pk_bf16_f32)
__device__ __forceinline__ v2bf cvt2(float a, float b) {
  return __builtin_amdgcn_cvt_pk_bf16_f32(a, b);
}
#else
__device__ __forceinline__ v2bf cvt2(float a, float b) {
  v2bf r;
  r[0] = (__bf16)a;
  r[1] = (__bf16)b;
  return r;
}
#endif

__device__ __forceinline__ unsigned pack2(float a, float b) {
  union { v2bf v; unsigned u; } t;
  t.v = cvt2(a, b);
  return t.u;
}

// Build a 16-element bf16 fragment from four f32 quads (K order used in this file:
// per-lane element i maps to K = i + 16*(lane/16) for A-fragments,
// K = i + 16*(lane/16) for B-fragments; callers pass quads in that K order).
__device__ __forceinline__ v16bf mk_frag(f4 x0, f4 x1, f4 x2, f4 x3) {
  union { v16bf v; v2bf p[8]; } r;
  r.p[0] = cvt2(x0.x, x0.y);
  r.p[1] = cvt2(x0.z, x0.w);
  r.p[2] = cvt2(x1.x, x1.y);
  r.p[3] = cvt2(x1.z, x1.w);
  r.p[4] = cvt2(x2.x, x2.y);
  r.p[5] = cvt2(x2.z, x2.w);
  r.p[6] = cvt2(x3.x, x3.y);
  r.p[7] = cvt2(x3.z, x3.w);
  return r.v;
}

// ---------------------------------------------------------------------------
// Kernel 0: one-time W_w f32 -> bf16 conversion.
// ---------------------------------------------------------------------------
__global__ __launch_bounds__(256)
void cvt_ww_kernel(const float* __restrict__ Ww, __bf16* __restrict__ WwBf) {
  const int i = (blockIdx.x * 256 + threadIdx.x) * 4;
  f4 v = *(const f4*)(Ww + i);
  u2 s;
  s.x = pack2(v.x, v.y);
  s.y = pack2(v.z, v.w);
  *(u2*)(WwBf + i) = s;
}

// ---------------------------------------------------------------------------
// Kernel 1: projection. One wave computes a 16-row strip of h (16 x 256),
// emits hT (bf16, [b][d][n], transposed for kernel-2 B-fragments) and w[b][n].
// A loads pipelined one k-step ahead; B fragments pipelined 2 tiles ahead.
// ---------------------------------------------------------------------------
__global__ __launch_bounds__(32)
void gat_proj_kernel(const float* __restrict__ hidden,
                     const __bf16* __restrict__ WwBf,
                     const float* __restrict__ Wb,
                     const float* __restrict__ aw,
                     const float* __restrict__ ab,
                     __bf16* __restrict__ hT,
                     float* __restrict__ wexp) {
  const int lane = threadIdx.x;
  const int half = lane >> 4;
  const int c    = lane & 15;
  const int b    = blockIdx.x / (N_ / 16);
  const int nBase = (blockIdx.x % (N_ / 16)) * 16;

  v8f acc[16];
  v8f zero = {0.f, 0.f, 0.f, 0.f, 0.f, 0.f, 0.f, 0.f};
#pragma unroll
  for (int t = 0; t < 16; ++t) acc[t] = zero;

  const float* Arow = hidden + (size_t)(b * N_ + nBase + c) * D_;
  const __bf16* Bbase = WwBf + (size_t)c * D_ + 16 * half;

  auto ldA = [&](int kb, f4 a[4]) {
    a[0] = *(const f4*)(Arow + kb + 16 * half);
    a[1] = *(const f4*)(Arow + kb + 16 * half + 4);
    a[2] = *(const f4*)(Arow + kb + 16 * half + 8);
    a[3] = *(const f4*)(Arow + kb + 16 * half + 12);
  };
  auto ldB = [&](int t, int kb) -> v16bf {
    return *(const v16bf*)(Bbase + (size_t)t * 16 * D_ + kb);
  };
  auto gemm_step = [&](int kb, const f4 a[4]) {
    v16bf af = mk_frag(a[0], a[1], a[2], a[3]);
    v16bf b0 = ldB(0, kb);
    v16bf b1 = ldB(1, kb);
#pragma unroll
    for (int t = 0; t < 16; ++t) {
      v16bf bn = b1;
      if (t < 14) bn = ldB(t + 2, kb);     // prefetch 2 tiles ahead
      acc[t] = __builtin_amdgcn_wmma_f32_16x16x32_bf16(
          false, af, false, b0, (short)0, acc[t], false, false);
      b0 = b1;
      b1 = bn;
    }
    pipeline_hint<0x020, 2>();             // VMEM-read groups, 2 trailing WMMAs
  };

  f4 ar[4], an[4];
  ldA(0, ar);
  for (int kb = 0; kb < D_; kb += 64) {
    ldA(kb + 32, an);                      // next A chunk during WMMAs
    gemm_step(kb, ar);
    if (kb + 64 < D_) ldA(kb + 64, ar);
    gemm_step(kb + 32, an);
  }

  // bias add + per-lane partial of f = h . a2   (a1 cancels in the softmax)
  float pf[8];
#pragma unroll
  for (int vg = 0; vg < 8; ++vg) pf[vg] = 0.f;
#pragma unroll
  for (int t = 0; t < 16; ++t) {
    float bias = Wb[16 * t + c];
    float a2c  = aw[D_ + 16 * t + c];
#pragma unroll
    for (int vg = 0; vg < 8; ++vg) {
      float hv = acc[t][vg] + bias;
      acc[t][vg] = hv;
      pf[vg] += a2c * hv;
    }
  }
  // reduce across the 16 lanes of each half (wave32 butterfly)
#pragma unroll
  for (int off = 1; off < 16; off <<= 1) {
#pragma unroll
    for (int vg = 0; vg < 8; ++vg) pf[vg] += __shfl_xor(pf[vg], off, 32);
  }
  if (c == 0) {
    float abv = ab[0];
#pragma unroll
    for (int vg = 0; vg < 8; ++vg)
      wexp[b * N_ + nBase + 8 * half + vg] = expf(pf[vg] + abv);
  }

  // store hT[b][d=16t+c][nBase + 8*half + 0..7] : 8 packed bf16 = one b128
#pragma unroll
  for (int t = 0; t < 16; ++t) {
    u4 p;
    p.x = pack2(acc[t][0], acc[t][1]);
    p.y = pack2(acc[t][2], acc[t][3]);
    p.z = pack2(acc[t][4], acc[t][5]);
    p.w = pack2(acc[t][6], acc[t][7]);
    *(u4*)(hT + (size_t)(b * D_ + 16 * t + c) * N_ + nBase + 8 * half) = p;
  }
}

// ---------------------------------------------------------------------------
// Kernel 2: h' = (S @ h) / (S @ 1), single pass over j in chunks of 32.
// Block = 4 waves (64 mask rows); hT triple-buffered via async-to-LDS with
// counted s_wait_asynccnt(8); mask/w pipelined one chunk ahead + prefetch;
// LDS B fragments pipelined 2 tiles ahead of the WMMAs (sched groups).
// ---------------------------------------------------------------------------
__global__ __launch_bounds__(32 * K2WAVES)
void gat_attn_kernel(const float* __restrict__ mask,
                     const __bf16* __restrict__ hT,
                     const float* __restrict__ wexp,
                     float* __restrict__ out) {
  __shared__ __align__(64) char smem[NBUF * BUFB];

  const int tid  = threadIdx.x;
  const int lane = tid & 31;
  const int wv   = tid >> 5;
  const int half = lane >> 4;
  const int c    = lane & 15;
  const int blocksPerBatch = N_ / (16 * K2WAVES);
  const int b       = blockIdx.x / blocksPerBatch;
  const int rowBase = (blockIdx.x % blocksPerBatch) * (16 * K2WAVES) + wv * 16;

  v8f acc[16];
  v8f dacc;
  v8f zero = {0.f, 0.f, 0.f, 0.f, 0.f, 0.f, 0.f, 0.f};
#pragma unroll
  for (int t = 0; t < 16; ++t) acc[t] = zero;
  dacc = zero;

  v16bf ones;
#pragma unroll
  for (int i = 0; i < 16; ++i) ones[i] = (__bf16)1.0f;

  const float*  mrow = mask + ((size_t)b * N_ + rowBase + c) * N_;
  const float*  wrow = wexp + (size_t)b * N_;
  const __bf16* hTb  = hT + (size_t)b * D_ * N_;

  // stage hT[:, jb..jb+31] (bf16) into LDS buffer `buf` (row-padded to 96B)
  auto stage = [&](int buf, int kc) {
    const int jb = kc * 32;
#pragma unroll
    for (int rr = 0; rr < 2; ++rr) {
      const int dd = tid + rr * (32 * K2WAVES);
      const __bf16* src = hTb + (size_t)dd * N_ + jb;
      char* dst = smem + buf * BUFB + dd * ROW_BYTES;
#ifdef USE_ASYNC_STAGE
#pragma unroll
      for (int q = 0; q < 4; ++q) {
        __builtin_amdgcn_global_load_async_to_lds_b128(
            (as1_i4*)(src + q * 8), (as3_i4*)(dst + q * 16), 0, 0);
      }
#else
      const u4* s4 = (const u4*)src;
      u4* d4 = (u4*)dst;
      d4[0] = s4[0]; d4[1] = s4[1]; d4[2] = s4[2]; d4[3] = s4[3];
#endif
    }
  };

  // A-side global loads for chunk kc (mask streamed NT + prefetch ahead)
  auto load_a = [&](int kc, f4 m[4], f4 w[4]) {
    const int jb = kc * 32;
    __builtin_prefetch(mrow + jb + 8 * 32, 0, 1);   // ~8 chunks (1KB) ahead
    m[0] = __builtin_nontemporal_load((const f4*)(mrow + jb + 16 * half));
    m[1] = __builtin_nontemporal_load((const f4*)(mrow + jb + 16 * half + 4));
    m[2] = __builtin_nontemporal_load((const f4*)(mrow + jb + 16 * half + 8));
    m[3] = __builtin_nontemporal_load((const f4*)(mrow + jb + 16 * half + 12));
    w[0] = *(const f4*)(wrow + jb + 16 * half);
    w[1] = *(const f4*)(wrow + jb + 16 * half + 4);
    w[2] = *(const f4*)(wrow + jb + 16 * half + 8);
    w[3] = *(const f4*)(wrow + jb + 16 * half + 12);
  };

  // 17 WMMAs against LDS buffer `buf`; B fragments pipelined 2 tiles ahead
  auto compute = [&](int buf, const f4* m, const f4* w) {
    v16bf af = mk_frag(m[0] * w[0], m[1] * w[1], m[2] * w[2], m[3] * w[3]);
    const char* base = smem + buf * BUFB + c * ROW_BYTES + half * 32;
    auto ldL = [&](int t) -> v16bf {
      return *(const v16bf*)(base + t * 16 * ROW_BYTES);
    };
    v16bf b0 = ldL(0);
    v16bf b1 = ldL(1);
#pragma unroll
    for (int t = 0; t < 16; ++t) {
      v16bf bn = b1;
      if (t < 14) bn = ldL(t + 2);
      acc[t] = __builtin_amdgcn_wmma_f32_16x16x32_bf16(
          false, af, false, b0, (short)0, acc[t], false, false);
      b0 = b1;
      b1 = bn;
    }
    dacc = __builtin_amdgcn_wmma_f32_16x16x32_bf16(
        false, af, false, ones, (short)0, dacc, false, false);
    pipeline_hint<0x100, 3>();             // DS-read groups, 3 trailing WMMAs
  };

  auto wrap3 = [](int x) { return x >= NBUF ? x - NBUF : x; };

  f4 mc[4], wc[4], mn[4], wn[4];
  stage(0, 0);
  stage(1, 1);
  load_a(0, mc, wc);
  WAIT_ASYNC(8);        // chunk0 complete (chunk1's 8 asyncs may stay in flight)
  __syncthreads();

  const int NCH = N_ / 32;   // 128, even
  int bc = 0;                // buffer holding chunk kc
  for (int kc = 0; kc < NCH; kc += 2) {
    if (kc + 2 < NCH) stage(wrap3(bc + 2), kc + 2);
    load_a(kc + 1, mn, wn);
    compute(bc, mc, wc);
    WAIT_ASYNC(8);
    __syncthreads();

    const int bc1 = wrap3(bc + 1);
    if (kc + 3 < NCH) stage(wrap3(bc1 + 2), kc + 3);
    if (kc + 2 < NCH) load_a(kc + 2, mc, wc);
    compute(bc1, mn, wn);
    WAIT_ASYNC(8);
    __syncthreads();

    bc = wrap3(bc1 + 1);
  }

  float invd[8];
#pragma unroll
  for (int vg = 0; vg < 8; ++vg) invd[vg] = 1.0f / dacc[vg];
#pragma unroll
  for (int t = 0; t < 16; ++t) {
#pragma unroll
    for (int vg = 0; vg < 8; ++vg) {
      const int row = rowBase + 8 * half + vg;
      __builtin_nontemporal_store(acc[t][vg] * invd[vg],
                                  out + ((size_t)b * N_ + row) * D_ + 16 * t + c);
    }
  }
}

// ---------------------------------------------------------------------------
extern "C" void kernel_launch(void* const* d_in, const int* in_sizes, int n_in,
                              void* d_out, int out_size, void* d_ws, size_t ws_size,
                              hipStream_t stream) {
  const float* hidden = (const float*)d_in[0];
  const float* mask   = (const float*)d_in[1];
  const float* Ww     = (const float*)d_in[2];
  const float* Wb     = (const float*)d_in[3];
  const float* aw     = (const float*)d_in[4];
  const float* ab     = (const float*)d_in[5];
  float* out = (float*)d_out;

  char* ws = (char*)d_ws;
  __bf16* hT   = (__bf16*)ws;                                   // 4 MB
  float*  wexp = (float*)(ws + (size_t)B_ * D_ * N_ * 2);       // +32 KB
  __bf16* WwBf = (__bf16*)(ws + (size_t)B_ * D_ * N_ * 2 + (size_t)B_ * N_ * 4); // +128 KB

  cvt_ww_kernel<<<(D_ * D_) / (256 * 4), 256, 0, stream>>>(Ww, WwBf);
  gat_proj_kernel<<<B_ * (N_ / 16), 32, 0, stream>>>(hidden, WwBf, Wb, aw, ab, hT, wexp);
  gat_attn_kernel<<<B_ * (N_ / (16 * K2WAVES)), 32 * K2WAVES, 0, stream>>>(mask, hT, wexp, out);
}
